// PhysBiformer_89026082112061
// MI455X (gfx1250) — 16-bit and 8-bit matrix operand order repaired and checked by reading, compiled, not run on hardware
//
#include <hip/hip_runtime.h>
#include <math.h>


typedef __attribute__((ext_vector_type(16))) _Float16 v16h;
typedef __attribute__((ext_vector_type(8)))  _Float16 v8h;
typedef __attribute__((ext_vector_type(8)))  float    v8f;
typedef __attribute__((ext_vector_type(8)))  int      v8i;
typedef __attribute__((ext_vector_type(4)))  float    f32x4;
typedef __attribute__((ext_vector_type(4)))  unsigned u32x4;
typedef __attribute__((ext_vector_type(4)))  int      i32x4;
typedef __attribute__((ext_vector_type(8)))  int      i32x8;

#if defined(__has_builtin)
#  if __has_builtin(__builtin_amdgcn_tensor_load_to_lds) && \
      __has_builtin(__builtin_amdgcn_s_wait_tensorcnt)
#    define PB_HAVE_TDM 1
#  endif
#endif
#ifndef PB_HAVE_TDM
#  define PB_HAVE_TDM 0
#endif
#if __has_include(<hip/amd_detail/amd_gfx1250_TDM.h>)
#  define PB_TDM_6ARG 1
#else
#  define PB_TDM_6ARG 0
#endif

#define ACT_NONE 0
#define ACT_RELU 1
#define ACT_ELU  2
#define ACT_LIF  3

static __device__ __forceinline__ float pb_act(float v, int act) {
  if (act == ACT_RELU) return v > 0.f ? v : 0.f;
  if (act == ACT_ELU)  return v > 0.f ? v : (__expf(v) - 1.f);
  if (act == ACT_LIF)  return (v * 0.5f - 1.f) >= 0.f ? 1.f : 0.f;
  return v;
}

union pb_frag16 { v16h v; v8h h[2]; };


__global__ void pb_cast_f32_f16(const float* __restrict__ src,
                                _Float16* __restrict__ dst, long n) {
  long i = (long)blockIdx.x * blockDim.x + threadIdx.x;
  if (i < n) dst[i] = (_Float16)src[i];
}

__global__ void pb_pack_conv_frag(const float* __restrict__ w, _Float16* __restrict__ wp,
                                  int O, int Ci, int KD, int KH, int KW, int K, int Kp) {
  int i = blockIdx.x * blockDim.x + threadIdx.x;
  if (i >= Kp * O) return;
  int e = i & 15; int t = i >> 4;
  int lane = t & 31; t >>= 5;
  int nt = O >> 4;
  int ot = t % nt, kt = t / nt;
  int k = kt * 32 + ((e < 8) ? ((lane >> 4) * 8 + e) : (16 + (lane >> 4) * 8 + (e - 8)));
  int o = ot * 16 + (lane & 15);
  _Float16 v = (_Float16)0.f;
  if (k < K) {
    int ci = k / (KD * KH * KW);
    int r  = k % (KD * KH * KW);
    int kd = r / (KH * KW); r %= KH * KW;
    int kh = r / KW, kw = r % KW;
    v = (_Float16)w[(((size_t)(o * Ci + ci) * KD + kd) * KH + kh) * KW + kw];
  }
  wp[i] = v;
}

__global__ void pb_pack_linear_frag(const float* __restrict__ w, _Float16* __restrict__ wp,
                                    int K, int N) {
  int i = blockIdx.x * blockDim.x + threadIdx.x;
  if (i >= K * N) return;
  int e = i & 15; int t = i >> 4;
  int lane = t & 31; t >>= 5;
  int nt = N >> 4;
  int ot = t % nt, kt = t / nt;
  int k = kt * 32 + ((e < 8) ? ((lane >> 4) * 8 + e) : (16 + (lane >> 4) * 8 + (e - 8)));
  int n = ot * 16 + (lane & 15);
  wp[i] = (_Float16)w[(size_t)k * N + n];
}

__global__ void pb_maxpool_hw(const _Float16* __restrict__ in, _Float16* __restrict__ out,
                              int NCD, int Ho, int Wo) {
  int i = blockIdx.x * blockDim.x + threadIdx.x;
  if (i >= NCD * Ho * Wo) return;
  int ncd = i / (Ho * Wo); int r = i % (Ho * Wo);
  int oh = r / Wo, ow = r % Wo;
  size_t b0 = ((size_t)ncd * (2 * Ho) + 2 * oh) * (2 * Wo) + 2 * ow;
  float m = fmaxf(fmaxf((float)in[b0], (float)in[b0 + 1]),
                  fmaxf((float)in[b0 + 2 * Wo], (float)in[b0 + 2 * Wo + 1]));
  out[i] = (_Float16)m;
}


template <int NB, int CI, int DI, int HI, int WI, int O, int DO, int HO, int WO,
          int SD, int SH, int SW, int PD, int PH, int PW,
          int KD, int KH, int KW, int KP, int ACT, int OUTTM>
__global__ __launch_bounds__(256) void pb_conv3d_wmma(
    const _Float16* __restrict__ in, const _Float16* __restrict__ wf,
    const float* __restrict__ bias,
    const float* __restrict__ bng, const float* __restrict__ bnb,
    const float* __restrict__ bnm, const float* __restrict__ bnv,
    void* __restrict__ out)
{
  constexpr int K   = CI * KD * KH * KW;
  constexpr int M   = NB * DO * HO * WO;
  constexpr int KHW = KH * KW, KDHW = KD * KH * KW;
  const int lane = threadIdx.x & 31;
  const int wv   = blockIdx.x * (blockDim.x >> 5) + (threadIdx.x >> 5);
  if (wv * 16 >= M) return;
  const int hi = lane >> 4, li = lane & 15;
  const int ncol = blockIdx.y * 16 + li;

  int row = wv * 16 + li;
  bool rowok = row < M;
  int rr = rowok ? row : 0;
  int n  = rr / (DO * HO * WO); int q = rr % (DO * HO * WO);
  int od = q / (HO * WO); q %= (HO * WO);
  int oh = q / WO, ow = q % WO;
  int id0 = od * SD - PD, ih0 = oh * SH - PH, iw0 = ow * SW - PW;

  const v16h* bt = (const v16h*)(wf + (size_t)blockIdx.y * 512);
  constexpr size_t tile_stride = (size_t)(O / 16) * 32;
  v8f acc = {};
#pragma unroll 1
  for (int kb = 0; kb < KP; kb += 32) {
    __builtin_prefetch(bt + tile_stride + lane, 0, 1);
    v16h a;
#pragma unroll
    for (int e = 0; e < 16; ++e) {
      int koff = (e < 8) ? (hi * 8 + e) : (16 + hi * 8 + (e - 8));
      int k = kb + koff;
      float v = 0.f;
      if (rowok && k < K) {
        int ci = k / KDHW; int kr = k % KDHW;
        int kd = kr / KHW; kr %= KHW;
        int kh = kr / KW, kw = kr % KW;
        int id = id0 + kd, ih = ih0 + kh, iw = iw0 + kw;
        if (id >= 0 && id < DI && ih >= 0 && ih < HI && iw >= 0 && iw < WI)
          v = (float)in[(((size_t)(n * CI + ci) * DI + id) * HI + ih) * WI + iw];
      }
      a[e] = (_Float16)v;
    }
    v16h b = bt[lane];
    bt += tile_stride;
    acc = __builtin_amdgcn_wmma_f32_16x16x32_f16(false, a, false, b, (short)0, acc,
                                                 false, false);
  }

  float scale = 1.f, shift = 0.f;
  if (bng) {
    float inv = rsqrtf(bnv[ncol] + 1e-5f);
    scale = bng[ncol] * inv;
    shift = bnb[ncol] - bnm[ncol] * scale;
  }
  float bsv = bias ? bias[ncol] : 0.f;
#pragma unroll
  for (int i = 0; i < 8; ++i) {
    int r = wv * 16 + hi * 8 + i;
    if (r >= M) continue;
    float v = (acc[i] + bsv) * scale + shift;
    v = pb_act(v, ACT);
    if (OUTTM) {
      ((float*)out)[(size_t)r * O + ncol] = v;
    } else {
      int nn = r / (DO * HO * WO); int t = r % (DO * HO * WO);
      int d2 = t / (HO * WO); t %= (HO * WO);
      int h2 = t / WO, w2 = t % WO;
      ((_Float16*)out)[(((size_t)(nn * O + ncol) * DO + d2) * HO + h2) * WO + w2] =
          (_Float16)v;
    }
  }
}


__global__ __launch_bounds__(256) void pb_gemm_wmma(
    const _Float16* __restrict__ A, const _Float16* __restrict__ Bw,
    const float* __restrict__ bias, int M, int K, int Nn,
    int act, int add_f32, void* __restrict__ out)
{
  const int lane = threadIdx.x & 31;
  const int hi = lane >> 4, li = lane & 15;
  const int mt = blockIdx.x * 8 + (threadIdx.x >> 5);
  if (mt * 16 >= M) return;
  const int ncol = blockIdx.y * 16 + li;
  const int row  = mt * 16 + li;
  const bool rowok = row < M;
  const int nt = Nn >> 4;
  const v16h* bt = (const v16h*)(Bw + (size_t)blockIdx.y * 512);
  const size_t tile_stride = (size_t)nt * 32;
  v8f acc = {};
  v8h z8 = {};
#pragma unroll 1
  for (int kb = 0; kb < K; kb += 32) {
    const v8h* pa = (const v8h*)(A + (size_t)(rowok ? row : 0) * K + kb);
    pb_frag16 ua;
    ua.h[0] = rowok ? pa[hi] : z8;
    ua.h[1] = rowok ? pa[hi + 2] : z8;
    v16h b = bt[lane];
    bt += tile_stride;
    acc = __builtin_amdgcn_wmma_f32_16x16x32_f16(false, ua.v, false, b, (short)0, acc,
                                                 false, false);
  }
  float bsv = bias ? bias[ncol] : 0.f;
#pragma unroll
  for (int i = 0; i < 8; ++i) {
    int r = mt * 16 + hi * 8 + i;
    if (r >= M) continue;
    float v = pb_act(acc[i] + bsv, act);
    if (add_f32) ((float*)out)[(size_t)r * Nn + ncol] += v;
    else ((_Float16*)out)[(size_t)r * Nn + ncol] = (_Float16)v;
  }
}


__global__ void pb_layernorm(const float* __restrict__ x, const float* __restrict__ g,
                             const float* __restrict__ b, _Float16* __restrict__ out,
                             int T, int C) {
  int t = blockIdx.x * blockDim.x + threadIdx.x;
  if (t >= T) return;
  const float* xr = x + (size_t)t * C;
  float m = 0.f;
  for (int c = 0; c < C; ++c) m += xr[c];
  m /= C;
  float var = 0.f;
  for (int c = 0; c < C; ++c) { float d = xr[c] - m; var += d * d; }
  var /= C;
  float inv = rsqrtf(var + 1e-5f);
  for (int c = 0; c < C; ++c)
    out[(size_t)t * C + c] = (_Float16)((xr[c] - m) * inv * g[c] + b[c]);
}


__global__ void pb_region_mean(const _Float16* __restrict__ qkv,
                               float* __restrict__ qm, float* __restrict__ km) {
  int i = blockIdx.x * blockDim.x + threadIdx.x;
  if (i >= 2 * 2 * 8 * 64) return;
  int which = i / (2 * 8 * 64); int r = i % (2 * 8 * 64);
  int bb = r / (8 * 64); r %= (8 * 64);
  int rg = r / 64, c = r % 64;
  int chan = which ? 64 + c : c;
  float s = 0.f;
  for (int m = 0; m < 80; ++m)
    s += (float)qkv[(size_t)(bb * 640 + rg * 80 + m) * 192 + chan];
  (which ? km : qm)[(bb * 8 + rg) * 64 + c] = s * (1.f / 80.f);
}

__global__ void pb_route_topk(const float* __restrict__ qm, const float* __restrict__ km,
                              int* __restrict__ idx) {
  int i = threadIdx.x;
  if (i >= 16) return;
  int bb = i / 8, rg = i % 8;
  float s[8];
  for (int j = 0; j < 8; ++j) {
    float d = 0.f;
    for (int c = 0; c < 64; ++c)
      d += qm[(bb * 8 + rg) * 64 + c] * km[(bb * 8 + j) * 64 + c];
    s[j] = d;
  }
  bool used[8] = {false, false, false, false, false, false, false, false};
  for (int t = 0; t < 4; ++t) {
    int best = 0; float bv = -3.0e38f;
    for (int j = 0; j < 8; ++j)
      if (!used[j] && s[j] > bv) { bv = s[j]; best = j; }
    used[best] = true;
    idx[(bb * 8 + rg) * 4 + t] = best;
  }
}


#define PB_KV_HALVES  (4 * 80 * 192)
#define PB_KV_BYTES   (PB_KV_HALVES * 2)
#define PB_ROW_HALVES (80 * 192)

__global__ __launch_bounds__(256) void pb_attn_wmma(
    const _Float16* __restrict__ qkv, const int* __restrict__ idx,
    _Float16* __restrict__ outbuf)
{
  extern __shared__ char pb_smem[];
  _Float16* kvlds = (_Float16*)pb_smem;
  float* logits = (float*)(pb_smem + PB_KV_BYTES);

  const int blk = blockIdx.x;
  const int bb = blk / 32;
  const int rg = (blk / 4) & 7;
  const int hh = blk & 3;
  const int lane = threadIdx.x & 31;
  const int wv = threadIdx.x >> 5;
  const int hi = lane >> 4, li = lane & 15;
  const int tokbase = bb * 640;
  const int* ridx = idx + (bb * 8 + rg) * 4;

#if PB_HAVE_TDM
  if (wv == 0) {
    unsigned long long ga =
        (unsigned long long)(const void*)(qkv + (size_t)tokbase * 192);
    unsigned ldsoff = (unsigned)(unsigned long long)(void*)kvlds;
    u32x4 g0;
    g0.x = 0x80000001u;
    g0.y = ldsoff;
    g0.z = (unsigned)(ga & 0xFFFFFFFFull);
    g0.w = (unsigned)((ga >> 32) & 0x01FFFFFFull) | 0x80000000u;
    i32x8 g1 = { 0x00010000,
                 (int)(PB_ROW_HALVES << 16),
                 0x00080000,
                 (int)(PB_ROW_HALVES << 16),
                 4,
                 PB_ROW_HALVES,
                 0, 0 };
    i32x4 g2; g2.x = (ridx[0] & 0xFFFF) | (ridx[1] << 16);
    g2.y = (ridx[2] & 0xFFFF) | (ridx[3] << 16); g2.z = 0; g2.w = 0;
    i32x4 g3 = {0, 0, 0, 0};
#if PB_TDM_6ARG
    i32x8 g4 = {0, 0, 0, 0, 0, 0, 0, 0};
    __builtin_amdgcn_tensor_load_to_lds(g0, g1, g2, g3, g4, 0);
#else
    __builtin_amdgcn_tensor_load_to_lds(g0, g1, g2, g3, 0);
#endif
    __builtin_amdgcn_s_wait_tensorcnt(0);
  }
#else
  for (int i = threadIdx.x; i < PB_KV_HALVES / 2; i += 256) {
    int j = i / (PB_ROW_HALVES / 2), r2 = i % (PB_ROW_HALVES / 2);
    ((unsigned*)kvlds)[i] =
        ((const unsigned*)(qkv + (size_t)(tokbase + ridx[j] * 80) * 192))[r2];
  }
#endif
  __syncthreads();

  for (int tile = wv; tile < 100; tile += 8) {
    int mi = tile / 20, ni = tile % 20;
    int m = mi * 16 + li;
    const v8h* qp =
        (const v8h*)(qkv + (size_t)(tokbase + rg * 80 + m) * 192 + hh * 16);
    v8h qv = qp[hi];
    v8i af = {};
    {
      int p0 = 0, p1 = 0;
#pragma unroll
      for (int by = 0; by < 4; ++by) {
        p0 |= (((float)qv[by] != 0.f) ? 1 : 0) << (8 * by);
        p1 |= (((float)qv[4 + by] != 0.f) ? 1 : 0) << (8 * by);
      }
      af[0] = p0; af[1] = p1;
    }
    int col = ni * 16 + li;
    int j = col / 80, mm = col % 80;
    const _Float16* krow = kvlds + (size_t)j * PB_ROW_HALVES + mm * 192 + 64 + hh * 16;
    v8h k0 = *(const v8h*)krow;
    v8h k1 = *(const v8h*)(krow + 8);
    v8i bf = {};
    {
      int p[4] = {0, 0, 0, 0};
#pragma unroll
      for (int by = 0; by < 4; ++by) {
        p[0] |= (((float)k0[by] != 0.f) ? 1 : 0) << (8 * by);
        p[1] |= (((float)k0[4 + by] != 0.f) ? 1 : 0) << (8 * by);
        p[2] |= (((float)k1[by] != 0.f) ? 1 : 0) << (8 * by);
        p[3] |= (((float)k1[4 + by] != 0.f) ? 1 : 0) << (8 * by);
      }
#pragma unroll
      for (int e = 0; e < 4; ++e) bf[e] = (e < 2) ? p[e + 2 * hi] : 0;
    }
    v8i cz = {};
    v8i d = __builtin_amdgcn_wmma_i32_16x16x64_iu8(false, af, false, bf, cz,
                                                   false, false);
#pragma unroll
    for (int i = 0; i < 8; ++i)
      logits[(mi * 16 + hi * 8 + i) * 320 + col] = (float)d[i] * 0.25f;
  }
  __syncthreads();

  for (int rrow = threadIdx.x; rrow < 80; rrow += blockDim.x) {
    float* Lr = logits + rrow * 320;
    float mx = -3.0e38f;
    for (int c = 0; c < 320; ++c) mx = fmaxf(mx, Lr[c]);
    float s = 0.f;
    for (int c = 0; c < 320; ++c) { float e = __expf(Lr[c] - mx); Lr[c] = e; s += e; }
    float inv = 1.f / s;
    for (int c = 0; c < 320; ++c) Lr[c] *= inv;
  }
  __syncthreads();

  if (wv < 5) {
    v8f acc = {};
#pragma unroll 1
    for (int kb = 0; kb < 320; kb += 32) {
      const float* Lr = logits + (wv * 16 + li) * 320 + kb + hi * 8;
      f32x4 c0 = *(const f32x4*)Lr;
      f32x4 c1 = *(const f32x4*)(Lr + 4);
      f32x4 c2 = *(const f32x4*)(Lr + 16);
      f32x4 c3 = *(const f32x4*)(Lr + 20);
      v16h a;
#pragma unroll
      for (int e = 0; e < 4; ++e) {
        a[e]      = (_Float16)c0[e];
        a[4 + e]  = (_Float16)c1[e];
        a[8 + e]  = (_Float16)c2[e];
        a[12 + e] = (_Float16)c3[e];
      }
      v16h b;
#pragma unroll
      for (int e = 0; e < 16; ++e) {
        int krow = kb + ((e < 8) ? (hi * 8 + e) : (16 + hi * 8 + (e - 8)));
        int j = krow / 80, mm = krow % 80;
        b[e] = kvlds[(size_t)j * PB_ROW_HALVES + mm * 192 + 128 + hh * 16 + li];
      }
      acc = __builtin_amdgcn_wmma_f32_16x16x32_f16(false, a, false, b, (short)0, acc,
                                                   false, false);
    }
#pragma unroll
    for (int i = 0; i < 8; ++i) {
      int m = wv * 16 + hi * 8 + i;
      outbuf[(size_t)(tokbase + rg * 80 + m) * 64 + hh * 16 + li] = (_Float16)acc[i];
    }
  }
}


__global__ void pb_tok_to_dhw_rep(const float* __restrict__ xs, _Float16* __restrict__ out) {
  int i = blockIdx.x * blockDim.x + threadIdx.x;
  if (i >= 2 * 64 * 80 * 16) return;
  int bb = i / (64 * 80 * 16); int r = i % (64 * 80 * 16);
  int c = r / (80 * 16); r %= (80 * 16);
  int od2 = r / 16, sp = r % 16;
  int tok = (od2 >> 1) * 16 + sp;
  out[i] = (_Float16)xs[((size_t)bb * 640 + tok) * 64 + c];
}

__global__ void pb_repeat_time(const _Float16* __restrict__ in, _Float16* __restrict__ out,
                               int NC, int Din, int HW) {
  int i = blockIdx.x * blockDim.x + threadIdx.x;
  if (i >= NC * Din * 2 * HW) return;
  int nc = i / (Din * 2 * HW); int r = i % (Din * 2 * HW);
  int t2 = r / HW, sp = r % HW;
  out[i] = in[((size_t)nc * Din + (t2 >> 1)) * HW + sp];
}

__global__ void pb_final_head(const _Float16* __restrict__ x, const float* __restrict__ lw,
                              const float* __restrict__ lb, float* __restrict__ out) {
  int i = blockIdx.x * blockDim.x + threadIdx.x;
  if (i >= 2 * 160) return;
  int bb = i / 160, t = i % 160;
  float acc = 0.f;
  for (int c = 0; c < 32; ++c) {
    float s = 0.f;
    for (int p = 0; p < 16; ++p)
      s += (float)x[(((size_t)(bb * 32 + c) * 160) + t) * 16 + p];
    acc += lw[c] * (s * (1.f / 16.f));
  }
  out[i] = acc + lb[0];
}


enum {
  IN_X = 0,
  IN_S0W, IN_S0B, IN_S0G, IN_S0BB, IN_S0M, IN_S0V,
  IN_S1W, IN_S1B, IN_S1G, IN_S1BB, IN_S1M, IN_S1V,
  IN_S2W, IN_S2B, IN_S2G, IN_S2BB, IN_S2M, IN_S2V,
  IN_PW, IN_PB,
  IN_BLK0
};
#define BLKI(b, j) (IN_BLK0 + (b) * 12 + (j))
#define IN_UP1W (IN_BLK0 + 36)
#define IN_UP1B (IN_UP1W + 1)
#define IN_UP1G (IN_UP1W + 2)
#define IN_UP1BB (IN_UP1W + 3)
#define IN_UP1M (IN_UP1W + 4)
#define IN_UP1V (IN_UP1W + 5)
#define IN_UP2W (IN_UP1W + 6)
#define IN_UP2B (IN_UP2W + 1)
#define IN_UP2G (IN_UP2W + 2)
#define IN_UP2BB (IN_UP2W + 3)
#define IN_UP2M (IN_UP2W + 4)
#define IN_UP2V (IN_UP2W + 5)
#define IN_LASTW (IN_UP2W + 6)
#define IN_LASTB (IN_LASTW + 1)

extern "C" void kernel_launch(void* const* d_in, const int* in_sizes, int n_in,
                              void* d_out, int out_size, void* d_ws, size_t ws_size,
                              hipStream_t stream) {
  (void)in_sizes; (void)n_in; (void)out_size; (void)ws_size;
  const float* fin = (const float*)d_in[IN_X];
  char* base = (char*)d_ws;
  size_t off = 0;
  auto alloc = [&](size_t bytes) -> char* {
    off = (off + 255) & ~(size_t)255;
    char* p = base + off;
    off += bytes;
    return p;
  };

  _Float16* bufX = (_Float16*)alloc((size_t)15728640 * 2);
  _Float16* big0 = (_Float16*)alloc((size_t)83886080 * 2);
  _Float16* big1 = (_Float16*)alloc((size_t)20971520 * 2);

  _Float16* w0p = (_Float16*)alloc((size_t)96 * 16 * 2);
  _Float16* w1p = (_Float16*)alloc((size_t)448 * 32 * 2);
  _Float16* w2p = (_Float16*)alloc((size_t)896 * 64 * 2);
  _Float16* wpp = (_Float16*)alloc((size_t)4096 * 64 * 2);
  _Float16* wu1 = (_Float16*)alloc((size_t)192 * 64 * 2);
  _Float16* wu2 = (_Float16*)alloc((size_t)192 * 32 * 2);
  _Float16 *wqkvp[3], *wop[3], *wf1p[3], *wf2p[3];
  for (int j = 0; j < 3; ++j) {
    wqkvp[j] = (_Float16*)alloc((size_t)64 * 192 * 2);
    wop[j]   = (_Float16*)alloc((size_t)64 * 64 * 2);
    wf1p[j]  = (_Float16*)alloc((size_t)64 * 256 * 2);
    wf2p[j]  = (_Float16*)alloc((size_t)256 * 64 * 2);
  }

  float*    xsnn  = (float*)alloc((size_t)1280 * 64 * 4);
  _Float16* lnbuf = (_Float16*)alloc((size_t)1280 * 64 * 2);
  _Float16* qkvb  = (_Float16*)alloc((size_t)1280 * 192 * 2);
  float*    qm    = (float*)alloc((size_t)16 * 64 * 4);
  float*    km    = (float*)alloc((size_t)16 * 64 * 4);
  int*      ridx  = (int*)alloc((size_t)16 * 4 * 4);
  _Float16* attno = (_Float16*)alloc((size_t)1280 * 64 * 2);
  _Float16* h1    = (_Float16*)alloc((size_t)1280 * 256 * 2);
  _Float16* rep1  = (_Float16*)alloc((size_t)2 * 64 * 80 * 16 * 2);
  _Float16* up1o  = (_Float16*)alloc((size_t)2 * 64 * 80 * 16 * 2);
  _Float16* rep2  = (_Float16*)alloc((size_t)2 * 64 * 160 * 16 * 2);
  _Float16* up2o  = (_Float16*)alloc((size_t)2 * 32 * 160 * 16 * 2);

  pb_cast_f32_f16<<<61440, 256, 0, stream>>>(fin, bufX, 15728640L);
  pb_pack_conv_frag<<<6, 256, 0, stream>>>((const float*)d_in[IN_S0W], w0p, 16, 3, 1, 5, 5, 75, 96);
  pb_pack_conv_frag<<<56, 256, 0, stream>>>((const float*)d_in[IN_S1W], w1p, 32, 16, 3, 3, 3, 432, 448);
  pb_pack_conv_frag<<<224, 256, 0, stream>>>((const float*)d_in[IN_S2W], w2p, 64, 32, 3, 3, 3, 864, 896);
  pb_pack_conv_frag<<<1024, 256, 0, stream>>>((const float*)d_in[IN_PW], wpp, 64, 64, 4, 4, 4, 4096, 4096);
  pb_pack_conv_frag<<<48, 256, 0, stream>>>((const float*)d_in[IN_UP1W], wu1, 64, 64, 3, 1, 1, 192, 192);
  pb_pack_conv_frag<<<24, 256, 0, stream>>>((const float*)d_in[IN_UP2W], wu2, 32, 64, 3, 1, 1, 192, 192);
  for (int j = 0; j < 3; ++j) {
    pb_pack_linear_frag<<<48, 256, 0, stream>>>((const float*)d_in[BLKI(j, 2)], wqkvp[j], 64, 192);
    pb_pack_linear_frag<<<16, 256, 0, stream>>>((const float*)d_in[BLKI(j, 4)], wop[j], 64, 64);
    pb_pack_linear_frag<<<64, 256, 0, stream>>>((const float*)d_in[BLKI(j, 8)], wf1p[j], 64, 256);
    pb_pack_linear_frag<<<64, 256, 0, stream>>>((const float*)d_in[BLKI(j, 10)], wf2p[j], 256, 64);
  }

  pb_conv3d_wmma<2, 3, 160, 128, 128, 16, 160, 128, 128,
                 1, 1, 1, 0, 2, 2, 1, 5, 5, 96, ACT_RELU, 0>
      <<<dim3(40960, 1), 256, 0, stream>>>(
      bufX, w0p, (const float*)d_in[IN_S0B],
      (const float*)d_in[IN_S0G], (const float*)d_in[IN_S0BB],
      (const float*)d_in[IN_S0M], (const float*)d_in[IN_S0V], big0);
  pb_maxpool_hw<<<81920, 256, 0, stream>>>(big0, big1, 2 * 16 * 160, 64, 64);

  pb_conv3d_wmma<2, 16, 160, 64, 64, 32, 160, 64, 64,
                 1, 1, 1, 1, 1, 1, 3, 3, 3, 448, ACT_RELU, 0>
      <<<dim3(10240, 2), 256, 0, stream>>>(
      big1, w1p, (const float*)d_in[IN_S1B],
      (const float*)d_in[IN_S1G], (const float*)d_in[IN_S1BB],
      (const float*)d_in[IN_S1M], (const float*)d_in[IN_S1V], big0);
  pb_maxpool_hw<<<40960, 256, 0, stream>>>(big0, bufX, 2 * 32 * 160, 32, 32);

  pb_conv3d_wmma<2, 32, 160, 32, 32, 64, 160, 32, 32,
                 1, 1, 1, 1, 1, 1, 3, 3, 3, 896, ACT_RELU, 0>
      <<<dim3(2560, 4), 256, 0, stream>>>(
      bufX, w2p, (const float*)d_in[IN_S2B],
      (const float*)d_in[IN_S2G], (const float*)d_in[IN_S2BB],
      (const float*)d_in[IN_S2M], (const float*)d_in[IN_S2V], big0);
  pb_maxpool_hw<<<20480, 256, 0, stream>>>(big0, big1, 2 * 64 * 160, 16, 16);

  pb_conv3d_wmma<2, 64, 160, 16, 16, 64, 40, 4, 4,
                 4, 4, 4, 0, 0, 0, 4, 4, 4, 4096, ACT_LIF, 1>
      <<<dim3(10, 4), 256, 0, stream>>>(
      big1, wpp, (const float*)d_in[IN_PB],
      nullptr, nullptr, nullptr, nullptr, xsnn);

  for (int j = 0; j < 3; ++j) {
    pb_layernorm<<<5, 256, 0, stream>>>(xsnn, (const float*)d_in[BLKI(j, 0)],
                                        (const float*)d_in[BLKI(j, 1)], lnbuf, 1280, 64);
    pb_gemm_wmma<<<dim3(10, 12), 256, 0, stream>>>(
        lnbuf, wqkvp[j], (const float*)d_in[BLKI(j, 3)], 1280, 64, 192,
        ACT_LIF, 0, qkvb);
    pb_region_mean<<<8, 256, 0, stream>>>(qkvb, qm, km);
    pb_route_topk<<<1, 32, 0, stream>>>(qm, km, ridx);
    pb_attn_wmma<<<64, 256, PB_KV_BYTES + 80 * 320 * sizeof(float), stream>>>(
        qkvb, ridx, attno);
    pb_gemm_wmma<<<dim3(10, 4), 256, 0, stream>>>(
        attno, wop[j], (const float*)d_in[BLKI(j, 5)], 1280, 64, 64,
        ACT_NONE, 1, xsnn);
    pb_layernorm<<<5, 256, 0, stream>>>(xsnn, (const float*)d_in[BLKI(j, 6)],
                                        (const float*)d_in[BLKI(j, 7)], lnbuf, 1280, 64);
    pb_gemm_wmma<<<dim3(10, 16), 256, 0, stream>>>(
        lnbuf, wf1p[j], (const float*)d_in[BLKI(j, 9)], 1280, 64, 256,
        ACT_LIF, 0, h1);
    pb_gemm_wmma<<<dim3(10, 4), 256, 0, stream>>>(
        h1, wf2p[j], (const float*)d_in[BLKI(j, 11)], 1280, 256, 64,
        ACT_LIF, 1, xsnn);
  }

  pb_tok_to_dhw_rep<<<640, 256, 0, stream>>>(xsnn, rep1);
  pb_conv3d_wmma<2, 64, 80, 4, 4, 64, 80, 4, 4,
                 1, 1, 1, 1, 0, 0, 3, 1, 1, 192, ACT_ELU, 0>
      <<<dim3(20, 4), 256, 0, stream>>>(
      rep1, wu1, (const float*)d_in[IN_UP1B],
      (const float*)d_in[IN_UP1G], (const float*)d_in[IN_UP1BB],
      (const float*)d_in[IN_UP1M], (const float*)d_in[IN_UP1V], up1o);
  pb_repeat_time<<<1280, 256, 0, stream>>>(up1o, rep2, 128, 80, 16);
  pb_conv3d_wmma<2, 64, 160, 4, 4, 32, 160, 4, 4,
                 1, 1, 1, 1, 0, 0, 3, 1, 1, 192, ACT_ELU, 0>
      <<<dim3(40, 2), 256, 0, stream>>>(
      rep2, wu2, (const float*)d_in[IN_UP2B],
      (const float*)d_in[IN_UP2G], (const float*)d_in[IN_UP2BB],
      (const float*)d_in[IN_UP2M], (const float*)d_in[IN_UP2V], up2o);
  pb_final_head<<<2, 256, 0, stream>>>(up2o, (const float*)d_in[IN_LASTW],
                                       (const float*)d_in[IN_LASTB], (float*)d_out);
}
